// DMPNNConv_64819646431793
// MI455X (gfx1250) — matrix operand order repaired and checked by reading, not compiled, not run on hardware
//
#include <hip/hip_runtime.h>

typedef __attribute__((ext_vector_type(16))) _Float16 v16h;
typedef __attribute__((ext_vector_type(8)))  _Float16 v8h;
typedef __attribute__((ext_vector_type(8)))  float    v8f;
typedef __attribute__((ext_vector_type(4)))  float    f4;

// D = 128 output features, K = 256 reduction (x:128 ++ h_e:128)
// Per workgroup: 256 threads = 8 waves, 128 rows (16 rows / wave).
// Per wave: 8 accumulator tiles (16x16 f32 = v8f each) covering all 128 cols.

// ---------------------------------------------------------------------------
// Pack W_i_w (f32 [128,256], row-major W[d,k]) into WMMA B-fragment f16 layout.
// frag f = kk*8 + t  (kk = k-tile 0..7 of 32, t = d-tile 0..7 of 16)
// B 32x16 16-bit layout: lane = n + 16*h, half j<8 <-> k = 8h+j, j>=8 <-> k = 16+8h+(j-8).
// ---------------------------------------------------------------------------
__global__ void k_pack(const float* __restrict__ Wi, _Float16* __restrict__ packed)
{
    int gid  = blockIdx.x * 256 + threadIdx.x;   // 0..2047
    int f    = gid >> 5;
    int lane = gid & 31;
    int kk = f >> 3, t = f & 7;
    int n  = lane & 15;
    int kb = kk * 32 + ((lane >> 4) << 3);
    #pragma unroll
    for (int j = 0; j < 16; ++j) {
        packed[(size_t)f * 512 + (size_t)lane * 16 + j] =
            (_Float16)Wi[(size_t)(t * 16 + n) * 256 + kb + j + ((j >> 3) << 3)];
    }
}

// ---------------------------------------------------------------------------
// GEMM core: acc[t] += A_tile(16x32) x B_tile(32x16) over 8 k-steps.
// A fragment built per-lane directly from global memory (NT: streamed):
//   lane l holds row m = l&15; lg = l>>4:
//   halves j=0..7  <-> k = 8*lg + (0..7); j=8..15 <-> k = 16 + 8*lg + (0..7)
// The two lane groups jointly consume whole 64B lines of each row.
// B fragments come from the packed 64KB table (WGP$/L2 resident, RT policy).
// A sched_group_barrier pipeline forces each k-step's 20 VMEM reads to issue
// one iteration ahead of the 8 WMMAs that consume them (software pipelining).
// ---------------------------------------------------------------------------
__device__ __forceinline__ void gemm_core(const float* __restrict__ x,
                                          const float* __restrict__ he,
                                          const v16h* __restrict__ Bp,
                                          int rowbase_w, int N, int lane,
                                          v8f* acc)
{
    const int m  = lane & 15;
    const int lg = lane >> 4;
    int rowg = rowbase_w + m;
    if (rowg >= N) rowg = N - 1;                 // clamp tail (masked later)
    const float* rowx = x  + (size_t)rowg * 128 + lg * 8;
    const float* rowe = he + (size_t)rowg * 128 + lg * 8;

    #pragma unroll
    for (int kk = 0; kk < 8; ++kk) {
        const float* src = (kk < 4) ? (rowx + kk * 32) : (rowe + (kk - 4) * 32);
        f4 f0 = __builtin_nontemporal_load((const f4*)(src + 0));
        f4 f1 = __builtin_nontemporal_load((const f4*)(src + 4));
        f4 f2 = __builtin_nontemporal_load((const f4*)(src + 16));
        f4 f3 = __builtin_nontemporal_load((const f4*)(src + 20));

        v16h b[8];
        #pragma unroll
        for (int t = 0; t < 8; ++t)
            b[t] = Bp[(size_t)(kk * 8 + t) * 32 + lane];

        v16h a;
        #pragma unroll
        for (int j = 0; j < 4; ++j) {
            a[j]      = (_Float16)f0[j];
            a[4 + j]  = (_Float16)f1[j];
            a[8 + j]  = (_Float16)f2[j];
            a[12 + j] = (_Float16)f3[j];
        }

        #pragma unroll
        for (int t = 0; t < 8; ++t)
            acc[t] = __builtin_amdgcn_wmma_f32_16x16x32_f16(
                         false, a, false, b[t], (short)0, acc[t], false, false);
    }

    // Pipeline spec for this scheduling region:
    //   per k-step: 20 VMEM reads (16 B-frag + 4 A), 8 WMMAs.
    //   Emit L0, L1, W0, L2, W1, ..., L7, W6, W7 so loads lead by one step.
    __builtin_amdgcn_sched_group_barrier(0x020, 20, 0);      // loads kk=0
    #pragma unroll
    for (int kk = 1; kk < 8; ++kk) {
        __builtin_amdgcn_sched_group_barrier(0x020, 20, 0);  // loads kk
        __builtin_amdgcn_sched_group_barrier(0x008, 8, 0);   // wmma  kk-1
    }
    __builtin_amdgcn_sched_group_barrier(0x008, 8, 0);       // wmma  kk=7
}

// ---------------------------------------------------------------------------
// Pass 1: h_n = relu(GEMM + b_i); deterministic per-WG reduction of
//         partial_m[d] = sum_rows s[row] * h_n[row, d]  (s[n] = sum_k bond[k,n])
// If SPILL: also store h_n as f16 [N,128] so pass 2 becomes elementwise.
// ---------------------------------------------------------------------------
template <bool SPILL>
__global__ __launch_bounds__(256) void k_pass1(
    const float* __restrict__ x, const float* __restrict__ he,
    const float* __restrict__ bond, const float* __restrict__ bi,
    const v16h* __restrict__ Bp, float* __restrict__ partials,
    _Float16* __restrict__ hn, int N, int Kb)
{
    __shared__ float sRed[8 * 128];
    const int tid  = threadIdx.x;
    const int w    = tid >> 5;
    const int lane = tid & 31;
    const int rowbase = blockIdx.x * 128 + w * 16;

    // s[row] for this wave's 16 rows (lanes 0..15), 0 for OOB rows.
    // Issued first so these HBM loads overlap the GEMM below.
    float s = 0.0f;
    {
        int rg = rowbase + (lane & 15);
        if (lane < 16 && rg < N) {
            for (int k = 0; k < Kb; ++k)
                s += __builtin_nontemporal_load(&bond[(size_t)k * N + rg]);
        }
    }

    v8f acc[8] = {};
    gemm_core(x, he, Bp, rowbase, N, lane, acc);

    const int n_lo = lane & 15;
    const int r_hi = (lane >> 4) << 3;

    // bias + relu (C layout: element (m = r + 8*(lane>>4), n = t*16 + (lane&15)))
    #pragma unroll
    for (int t = 0; t < 8; ++t) {
        float bv = bi[t * 16 + n_lo];
        #pragma unroll
        for (int r = 0; r < 8; ++r) {
            float h = acc[t][r] + bv;
            acc[t][r] = h > 0.0f ? h : 0.0f;
            if (SPILL) {
                int rg = rowbase + r + r_hi;
                if (rg < N)
                    __builtin_nontemporal_store((_Float16)acc[t][r],
                        &hn[(size_t)rg * 128 + t * 16 + n_lo]);
            }
        }
    }

    // lane needs s[row = r + 8*(lane>>4)] -> source lane r + 8*(lane>>4)
    float sv[8];
    #pragma unroll
    for (int r = 0; r < 8; ++r) sv[r] = __shfl(s, r + ((lane >> 4) << 3), 32);

    #pragma unroll
    for (int t = 0; t < 8; ++t) {
        float p = 0.0f;
        #pragma unroll
        for (int r = 0; r < 8; ++r) p += sv[r] * acc[t][r];
        p += __shfl_xor(p, 16, 32);             // combine row halves
        if (lane < 16) sRed[w * 128 + t * 16 + lane] = p;
    }
    __syncthreads();
    if (tid < 128) {
        float msum = 0.0f;
        #pragma unroll
        for (int w2 = 0; w2 < 8; ++w2) msum += sRed[w2 * 128 + tid];
        partials[(size_t)blockIdx.x * 128 + tid] = msum;
    }
}

// ---------------------------------------------------------------------------
// Combine: m[d] = fixed-order sum of partials (8 fixed strided slices + tree),
//          c[d] = W_m[d,:] @ m + b_m[d].   1 block x 1024 threads.
// ---------------------------------------------------------------------------
__global__ __launch_bounds__(1024) void k_combine(
    const float* __restrict__ partials, int numWG,
    const float* __restrict__ Wm, const float* __restrict__ bm,
    float* __restrict__ cvec)
{
    __shared__ float sg[8][128];
    __shared__ float sm[128];
    const int tid = threadIdx.x;
    const int d   = tid & 127;
    const int grp = tid >> 7;                    // 0..7
    float m = 0.0f;
    for (int g = grp; g < numWG; g += 8)
        m += partials[(size_t)g * 128 + d];
    sg[grp][d] = m;
    __syncthreads();
    if (tid < 128) {
        float t = 0.0f;
        #pragma unroll
        for (int g = 0; g < 8; ++g) t += sg[g][d];   // fixed order
        sm[d] = t;
    }
    __syncthreads();
    if (tid < 128) {
        float c = bm[d];
        for (int j = 0; j < 128; ++j) c += Wm[(size_t)d * 128 + j] * sm[j];
        cvec[d] = c;
    }
}

// ---------------------------------------------------------------------------
// Pass 2 (fallback): recompute h_n, fuse h = relu(relu(h_n)+c), store (NT).
// ---------------------------------------------------------------------------
__global__ __launch_bounds__(256) void k_pass2(
    const float* __restrict__ x, const float* __restrict__ he,
    const float* __restrict__ bi, const v16h* __restrict__ Bp,
    const float* __restrict__ cvec, float* __restrict__ out, int N)
{
    const int tid  = threadIdx.x;
    const int w    = tid >> 5;
    const int lane = tid & 31;
    const int rowbase = blockIdx.x * 128 + w * 16;

    v8f acc[8] = {};
    gemm_core(x, he, Bp, rowbase, N, lane, acc);

    const int n_lo = lane & 15;
    const int r_hi = (lane >> 4) << 3;
    #pragma unroll
    for (int t = 0; t < 8; ++t) {
        float bv = bi[t * 16 + n_lo];
        float cv = cvec[t * 16 + n_lo];
        #pragma unroll
        for (int r = 0; r < 8; ++r) {
            float hn = acc[t][r] + bv;
            hn = hn > 0.0f ? hn : 0.0f;
            float h = hn + cv;
            h = h > 0.0f ? h : 0.0f;
            int rg = rowbase + r + r_hi;
            if (rg < N)
                __builtin_nontemporal_store(h, &out[(size_t)rg * 128 + t * 16 + n_lo]);
        }
    }
}

// ---------------------------------------------------------------------------
// Finalize (spill path): out[n,d] = relu(h_n_f16[n,d] + c[d]), 8 elems/thread.
// ---------------------------------------------------------------------------
__global__ __launch_bounds__(256) void k_finalize(
    const _Float16* __restrict__ hn, const float* __restrict__ cvec,
    float* __restrict__ out, long total8)
{
    long idx = (long)blockIdx.x * 256 + threadIdx.x;
    if (idx >= total8) return;
    v8h h = __builtin_nontemporal_load((const v8h*)hn + idx);
    int colbase = ((int)idx << 3) & 127;
    f4 o0, o1;
    #pragma unroll
    for (int j = 0; j < 4; ++j) {
        float a = (float)h[j]     + cvec[colbase + j];
        float b = (float)h[4 + j] + cvec[colbase + 4 + j];
        o0[j] = a > 0.0f ? a : 0.0f;
        o1[j] = b > 0.0f ? b : 0.0f;
    }
    __builtin_nontemporal_store(o0, (f4*)(out + idx * 8));
    __builtin_nontemporal_store(o1, (f4*)(out + idx * 8 + 4));
}

// ---------------------------------------------------------------------------
extern "C" void kernel_launch(void* const* d_in, const int* in_sizes, int n_in,
                              void* d_out, int out_size, void* d_ws, size_t ws_size,
                              hipStream_t stream)
{
    const float* x    = (const float*)d_in[0];
    const float* he   = (const float*)d_in[1];
    const float* bond = (const float*)d_in[2];
    const float* Wi   = (const float*)d_in[3];
    const float* bi   = (const float*)d_in[4];
    const float* Wm   = (const float*)d_in[5];
    const float* bm   = (const float*)d_in[6];
    float* out = (float*)d_out;

    const int N  = in_sizes[0] / 128;
    const int Kb = in_sizes[2] / N;
    const int numWG = (N + 127) >> 7;

    char* ws = (char*)d_ws;
    _Float16* packed = (_Float16*)ws;                              // 64 KB
    float* partials  = (float*)(ws + 65536);                       // numWG*512 B
    size_t off_cvec  = 65536 + (size_t)numWG * 512;
    float* cvec      = (float*)(ws + off_cvec);                    // 512 B
    size_t off_hn    = off_cvec + 512;
    _Float16* hn     = (_Float16*)(ws + off_hn);                   // N*256 B
    const bool spill = ws_size >= off_hn + (size_t)N * 256;

    k_pack<<<8, 256, 0, stream>>>(Wi, packed);
    if (spill) {
        k_pass1<true><<<numWG, 256, 0, stream>>>(x, he, bond, bi,
            (const v16h*)packed, partials, hn, N, Kb);
        k_combine<<<1, 1024, 0, stream>>>(partials, numWG, Wm, bm, cvec);
        long total8 = (long)N * 16;
        int blocks = (int)((total8 + 255) / 256);
        k_finalize<<<blocks, 256, 0, stream>>>(hn, cvec, out, total8);
    } else {
        k_pass1<false><<<numWG, 256, 0, stream>>>(x, he, bond, bi,
            (const v16h*)packed, partials, nullptr, N, Kb);
        k_combine<<<1, 1024, 0, stream>>>(partials, numWG, Wm, bm, cvec);
        k_pass2<<<numWG, 256, 0, stream>>>(x, he, bi, (const v16h*)packed,
                                           cvec, out, N);
    }
}
